// Attend_58815282151496
// MI455X (gfx1250) — matrix operand order repaired and checked by reading, not compiled, not run on hardware
//
#include <hip/hip_runtime.h>
#include <hip/hip_bf16.h>
#include <cfloat>

typedef __attribute__((ext_vector_type(16))) _Float16 v16h;
typedef __attribute__((ext_vector_type(8)))  _Float16 v8h;
typedef __attribute__((ext_vector_type(2)))  _Float16 v2h;
typedef __attribute__((ext_vector_type(8)))  float    v8f;

#define ATT_B 2
#define ATT_H 16
#define ATT_N 2048
#define ATT_D 64
#define ROWS_PER_BLOCK 64                 // 4 waves x 16 query rows
#define ITILES (ATT_N / ROWS_PER_BLOCK)   // 32

#if __has_builtin(__builtin_amdgcn_exp2f)
#define EXP2F(x) __builtin_amdgcn_exp2f(x)
#else
#define EXP2F(x) exp2f(x)
#endif

// load 8 consecutive floats (two float4) into t[0..7]
__device__ __forceinline__ void load8(const float* p, float* t) {
  float4 a = ((const float4*)p)[0];
  float4 b = ((const float4*)p)[1];
  t[0]=a.x; t[1]=a.y; t[2]=a.z; t[3]=a.w;
  t[4]=b.x; t[5]=b.y; t[6]=b.z; t[7]=b.w;
}

// --- lane reductions in the VALU pipe (no ds_bpermute) -------------------
template <int CTRL>
__device__ __forceinline__ float fmax_dpp(float v) {
  int x = __builtin_amdgcn_update_dpp(0, __builtin_bit_cast(int, v),
                                      CTRL, 0xF, 0xF, true);
  return fmaxf(v, __builtin_bit_cast(float, x));
}
// max over each 16-lane row: xor1, xor2, xor7(row_half_mirror), xor15(row_mirror)
__device__ __forceinline__ float redmax16(float v) {
  v = fmax_dpp<0xB1>(v);    // quad_perm(1,0,3,2)
  v = fmax_dpp<0x4E>(v);    // quad_perm(2,3,0,1)
  v = fmax_dpp<0x141>(v);   // row_half_mirror
  v = fmax_dpp<0x140>(v);   // row_mirror
  return v;
}
// v + (value from lane ^ 16) via permlanex16 with identity selects
__device__ __forceinline__ float addx16(float v) {
  int s = __builtin_bit_cast(int, v);
  int x = __builtin_amdgcn_permlanex16(s, s, 0x76543210, 0xFEDCBA98, false, false);
  return v + __builtin_bit_cast(float, x);
}

__global__ __launch_bounds__(128)
void attend_l2_wmma(const float* __restrict__ Q, const float* __restrict__ K,
                    const float* __restrict__ V, const int* __restrict__ Mask,
                    float* __restrict__ Out)
{
  // padded stride 40 halves = 80B rows: 16B-aligned chunks + bank spreading
  __shared__ _Float16 VT[ATT_D][40];   // V^T tile (f16): VT[d][j] = V[jb+j][d]
  __shared__ _Float16 PB[4][16][40];   // per-wave P tile (C-layout -> A-layout)
  __shared__ float    QS[4][16];       // per-wave row |q|^2

  const int lane = threadIdx.x & 31;
  const int wave = threadIdx.x >> 5;
  const int n    = lane & 15;          // column index inside 16-wide tile
  const int h    = lane >> 4;          // wave half

  const int bh    = blockIdx.x / ITILES;
  const int itile = blockIdx.x % ITILES;
  const int b     = bh / ATT_H;
  const size_t base = (size_t)bh * ATT_N * ATT_D;
  const float* Qb = Q + base;
  const float* Kb = K + base;
  const float* Vb = V + base;
  const int*   Mb = Mask + (size_t)b * ATT_N;
  float*       Ob = Out + base;

  const int i0 = itile * ROWS_PER_BLOCK + wave * 16;

  const float L2E = 1.44269504088896340736f;      // log2(e)
  const float TS2 = 0.25f * 1.44269504088896340736f; // 2*D^-0.5 * log2(e)
  const float MASK_BIAS = 3.0e38f;                // huge log2-domain bias

  // ---- Q fragments (16x32 f16 A-layout x2 for d=0..63) + |q|^2 ----------
  // A layout: lane m = lane%16; elem e<8 -> d = e + 8h; e>=8 -> d = e+8 + 8h
  v16h a0, a1;
  float qsql = 0.f;
  {
    const float* qrow = Qb + (size_t)(i0 + n) * ATT_D;
    float t0[16], t1[16];
    load8(qrow +      8*h, t0);
    load8(qrow + 16 + 8*h, t0 + 8);
    load8(qrow + 32 + 8*h, t1);
    load8(qrow + 48 + 8*h, t1 + 8);
    #pragma unroll
    for (int e = 0; e < 16; ++e) {
      qsql += t0[e]*t0[e] + t1[e]*t1[e];
      a0[e] = (_Float16)t0[e];
      a1[e] = (_Float16)t1[e];
    }
  }
  qsql = addx16(qsql);                  // combine halves -> full 64-d sum
  if (h == 0) QS[wave][n] = qsql;
  __syncthreads();
  float qsq2[8];                        // log2-scaled |q|^2, C-layout rows
  #pragma unroll
  for (int r = 0; r < 8; ++r) qsq2[r] = QS[wave][r + 8*h] * L2E;

  // all-ones B fragment: rowsum(P) = P @ ones lands in C layout
  v16h ones;
  #pragma unroll
  for (int e = 0; e < 16; ++e) ones[e] = (_Float16)1.0f;

  // ---- online softmax state (log2 domain) -------------------------------
  float mrun[8], lrun[8];
  v8f o0 = {}, o1 = {}, o2 = {}, o3 = {};
  #pragma unroll
  for (int r = 0; r < 8; ++r) { mrun[r] = -FLT_MAX; lrun[r] = 0.f; }

  for (int jb = 0; jb < ATT_N; jb += 32) {
    // ---- cooperative stage of V^T tile (f32 -> f16, packed b32) --------
    __syncthreads();                    // previous tile's reads done
    {
      const int jp = (threadIdx.x & 15) * 2;   // even local key column
      const int dc = (threadIdx.x >> 4) * 8;   // 8-wide d chunk
      const float* vr0 = Vb + (size_t)(jb + jp) * ATT_D + dc;
      const float* vr1 = vr0 + ATT_D;
      if (jb + 32 < ATT_N) __builtin_prefetch(vr0 + 32*ATT_D, 0, 1);
      float t0[8], t1[8];
      load8(vr0, t0);
      load8(vr1, t1);
      #pragma unroll
      for (int e = 0; e < 8; ++e) {
        v2h pk; pk[0] = (_Float16)t0[e]; pk[1] = (_Float16)t1[e];
        *(v2h*)&VT[dc + e][jp] = pk;     // one ds_store_b32, v_cvt_pk_f16_f32
      }
    }
    __syncthreads();

    // ---- build BOTH K-tile fragments first -----------------------------
    v16h bf[4];                          // b00,b01 (tile0), b10,b11 (tile1)
    float ksa[2];
    #pragma unroll
    for (int t = 0; t < 2; ++t) {
      const int j = jb + 16*t + n;
      const float* krow = Kb + (size_t)j * ATT_D;
      if (jb + 32 < ATT_N) __builtin_prefetch(krow + 32*ATT_D, 0, 1);
      // B layout (K^T, 32x16): lane n holds key row j; elem e<8 -> d = e + 8h; e>=8 -> d = e+8 + 8h
      float tk0[16], tk1[16];
      load8(krow +      8*h, tk0);
      load8(krow + 16 + 8*h, tk0 + 8);
      load8(krow + 32 + 8*h, tk1);
      load8(krow + 48 + 8*h, tk1 + 8);
      v16h b0, b1;
      float ks = 0.f;
      #pragma unroll
      for (int e = 0; e < 16; ++e) {
        ks += tk0[e]*tk0[e] + tk1[e]*tk1[e];
        b0[e] = (_Float16)tk0[e];
        b1[e] = (_Float16)tk1[e];
      }
      ks = addx16(ks);                  // full-row |k|^2 at column lane
      // log2-scale + fold key mask in as a huge bias (one cndmask per tile)
      ksa[t] = fmaf(ks, L2E, (Mb[j] > 0) ? MASK_BIAS : 0.0f);
      bf[2*t]     = b0;
      bf[2*t + 1] = b1;
    }

    // ---- 4 QK WMMAs interleaved across 2 independent accumulators ------
    v8f c0 = {}, c1 = {};
    c0 = __builtin_amdgcn_wmma_f32_16x16x32_f16(false, a0, false, bf[0], (short)0, c0, false, false);
    c1 = __builtin_amdgcn_wmma_f32_16x16x32_f16(false, a0, false, bf[2], (short)0, c1, false, false);
    c0 = __builtin_amdgcn_wmma_f32_16x16x32_f16(false, a1, false, bf[1], (short)0, c0, false, false);
    c1 = __builtin_amdgcn_wmma_f32_16x16x32_f16(false, a1, false, bf[3], (short)0, c1, false, false);

    // scores in log2 domain: s = log2e*(2*scale*qk - |q|^2 - |k|^2)
    float s0[8], s1[8];
    #pragma unroll
    for (int r = 0; r < 8; ++r) {
      s0[r] = fmaf(TS2, c0[r], -(qsq2[r] + ksa[0]));
      s1[r] = fmaf(TS2, c1[r], -(qsq2[r] + ksa[1]));
    }

    // ---- online softmax: max via DPP, exp2, stage P into LDS -----------
    float alpha[8];
    #pragma unroll
    for (int r = 0; r < 8; ++r) {
      float mx   = redmax16(fmaxf(s0[r], s1[r]));
      float mnew = fmaxf(mrun[r], mx);
      alpha[r]   = EXP2F(mrun[r] - mnew);
      mrun[r]    = mnew;
      float p0 = EXP2F(s0[r] - mnew);
      float p1 = EXP2F(s1[r] - mnew);
      _Float16* pr = &PB[wave][r + 8*h][0];
      pr[n]      = (_Float16)p0;
      pr[16 + n] = (_Float16)p1;
      o0[r] *= alpha[r]; o1[r] *= alpha[r]; o2[r] *= alpha[r]; o3[r] *= alpha[r];
    }
    __asm volatile("s_wait_dscnt 0" ::: "memory");  // wave-local LDS RAW

    v16h pa;
    {
      const _Float16* pr = &PB[wave][n][0];
      union { v16h v; v8h q[2]; } u;
      u.q[0] = *(const v8h*)(pr +      8*h);
      u.q[1] = *(const v8h*)(pr + 16 + 8*h);
      pa = u.v;
    }

    // ---- l update via WMMA: rowsum(P) = P @ ones -----------------------
    {
      v8f csum = {};
      csum = __builtin_amdgcn_wmma_f32_16x16x32_f16(false, pa, false, ones, (short)0, csum, false, false);
      #pragma unroll
      for (int r = 0; r < 8; ++r)
        lrun[r] = lrun[r] * alpha[r] + csum[r];
    }

    // ---- O += P @ V (B fragments from transposed V in LDS) -------------
    #pragma unroll
    for (int nb = 0; nb < 4; ++nb) {
      const _Float16* vtr = &VT[16*nb + n][0];   // B elem e<8 -> key 8h + e; e>=8 -> key 16 + 8h + (e-8)
      union { v16h v; v8h q[2]; } u;
      u.q[0] = *(const v8h*)(vtr + 8*h);
      u.q[1] = *(const v8h*)(vtr + 16 + 8*h);
      v8f& o = (nb == 0) ? o0 : (nb == 1) ? o1 : (nb == 2) ? o2 : o3;
      o = __builtin_amdgcn_wmma_f32_16x16x32_f16(false, pa, false, u.v, (short)0, o, false, false);
    }
  }

  // ---- epilogue: normalize and store -----------------------------------
  #pragma unroll
  for (int r = 0; r < 8; ++r) {
    const float inv = 1.0f / lrun[r];
    float* orow = Ob + (size_t)(i0 + r + 8*h) * ATT_D;
    orow[ 0 + n] = o0[r] * inv;
    orow[16 + n] = o1[r] * inv;
    orow[32 + n] = o2[r] * inv;
    orow[48 + n] = o3[r] * inv;
  }
}

extern "C" void kernel_launch(void* const* d_in, const int* in_sizes, int n_in,
                              void* d_out, int out_size, void* d_ws, size_t ws_size,
                              hipStream_t stream) {
  const float* q    = (const float*)d_in[0];
  const float* k    = (const float*)d_in[1];
  const float* v    = (const float*)d_in[2];
  const int*   mask = (const int*)d_in[3];
  float* out = (float*)d_out;
  (void)in_sizes; (void)n_in; (void)out_size; (void)d_ws; (void)ws_size;

  dim3 grid(ATT_B * ATT_H * ITILES);   // 1024 workgroups
  dim3 block(128);                     // 4 waves (wave32)
  attend_l2_wmma<<<grid, block, 0, stream>>>(q, k, v, mask, out);
}
